// FeedForwardQuantum_65481071399611
// MI455X (gfx1250) — matrix operand order repaired and checked by reading, not compiled, not run on hardware
//
#include <hip/hip_runtime.h>

// ---- vector types matching the CDNA5 WMMA builtin signatures ----
typedef __attribute__((ext_vector_type(8)))  float  v8f;
typedef __attribute__((ext_vector_type(4)))  float  v4f;
typedef __attribute__((ext_vector_type(16))) __bf16 v16bf;
typedef __attribute__((ext_vector_type(8)))  __bf16 v8bf;

static __device__ __forceinline__ v16bf cat8(v8bf lo, v8bf hi) {
  return __builtin_shufflevector(lo, hi, 0, 1, 2, 3, 4, 5, 6, 7,
                                         8, 9, 10, 11, 12, 13, 14, 15);
}

// ---------------------------------------------------------------------------
// Pass 1: q[i] = cos(x[i]) * cos(theta[i % D]) stored as bf16, 8 elems/thread.
// D is a multiple of 8, so (t*8) % D + j never crosses the theta row.
// ---------------------------------------------------------------------------
__global__ __launch_bounds__(256) void quantum_pack_bf16(
    const float* __restrict__ x, const float* __restrict__ theta,
    __bf16* __restrict__ q, long long n8, int D) {
  long long t = (long long)blockIdx.x * blockDim.x + threadIdx.x;
  if (t >= n8) return;
  long long base = t * 8;
  int dbase = (int)(base % (long long)D);
  v4f x0 = *reinterpret_cast<const v4f*>(x + base);
  v4f x1 = *reinterpret_cast<const v4f*>(x + base + 4);
  v4f t0 = *reinterpret_cast<const v4f*>(theta + dbase);
  v4f t1 = *reinterpret_cast<const v4f*>(theta + dbase + 4);
  v8bf r;
#pragma unroll
  for (int j = 0; j < 4; ++j) {
    r[j]     = (__bf16)(__cosf(x0[j]) * __cosf(t0[j]));
    r[j + 4] = (__bf16)(__cosf(x1[j]) * __cosf(t1[j]));
  }
  *reinterpret_cast<v8bf*>(q + base) = r;
}

// ---------------------------------------------------------------------------
// fp32 -> bf16 weight conversion, 8 elems/thread.
// ---------------------------------------------------------------------------
__global__ __launch_bounds__(256) void cvt_f32_to_bf16(
    const float* __restrict__ src, __bf16* __restrict__ dst, long long n8) {
  long long t = (long long)blockIdx.x * blockDim.x + threadIdx.x;
  if (t >= n8) return;
  long long base = t * 8;
  v4f s0 = *reinterpret_cast<const v4f*>(src + base);
  v4f s1 = *reinterpret_cast<const v4f*>(src + base + 4);
  v8bf r;
#pragma unroll
  for (int j = 0; j < 4; ++j) {
    r[j]     = (__bf16)s0[j];
    r[j + 4] = (__bf16)s1[j];
  }
  *reinterpret_cast<v8bf*>(dst + base) = r;
}

// ---------------------------------------------------------------------------
// Fragment set for one 64x64 tile K-step (4 A frags of 16x32, 4 B of 32x16).
// ---------------------------------------------------------------------------
struct Frags {
  v16bf a[4];
  v16bf b[4];
};

static __device__ __forceinline__ void load_frags(
    Frags& f, const __bf16* const (&aPtr)[4], const __bf16* const (&bPtr)[4],
    int k0) {
#pragma unroll
  for (int i = 0; i < 4; ++i)
    f.a[i] = cat8(*reinterpret_cast<const v8bf*>(aPtr[i] + k0),
                  *reinterpret_cast<const v8bf*>(aPtr[i] + k0 + 8));
#pragma unroll
  for (int j = 0; j < 4; ++j)
    f.b[j] = cat8(*reinterpret_cast<const v8bf*>(bPtr[j] + k0),
                  *reinterpret_cast<const v8bf*>(bPtr[j] + k0 + 8));
}

static __device__ __forceinline__ void mma_tile(v8f (&acc)[4][4],
                                                const Frags& f) {
#pragma unroll
  for (int i = 0; i < 4; ++i)
#pragma unroll
    for (int j = 0; j < 4; ++j)
      acc[i][j] = __builtin_amdgcn_wmma_f32_16x16x32_bf16(
          false, f.a[i], false, f.b[j], (short)0, acc[i][j], false, false);
}

// ---------------------------------------------------------------------------
// C[M,N] = act(A[M,K] @ Bm[N,K]^T)   (torch-Linear layout: Bm is [out,in])
// One wave32 per 64x64 C tile: 4x4 fragments of v_wmma_f32_16x16x32_bf16.
// 256-thread block = 8 waves = 8 tiles (consecutive in N -> A rows shared in
// WGP caches). Requires M%64==0, N%64==0, K%64==0.
//
// K loop is 2x unrolled with double-buffered fragment registers so the
// global_load_b128s of step k+32 never write the source VGPRs of the
// in-flight WMMAs of step k (avoids the WAR interlock at the loop boundary).
// ---------------------------------------------------------------------------
template <bool RELU, typename OUT_T>
__global__ __launch_bounds__(256) void gemm_bf16_wmma(
    const __bf16* __restrict__ A, const __bf16* __restrict__ Bm,
    OUT_T* __restrict__ C, int M, int N, int K) {
  const int lane = threadIdx.x & 31;
  const int wave = threadIdx.x >> 5;
  const int col  = lane & 15;
  const int hf   = lane >> 4;

  const int tilesN = N >> 6;
  const int tile   = blockIdx.x * 8 + wave;
  const long long m0 = (long long)(tile / tilesN) * 64;
  const long long n0 = (long long)(tile % tilesN) * 64;

  const __bf16* aPtr[4];
  const __bf16* bPtr[4];
#pragma unroll
  for (int i = 0; i < 4; ++i)
    aPtr[i] = A + (m0 + (long long)(i * 16 + col)) * (long long)K + hf * 16;
#pragma unroll
  for (int j = 0; j < 4; ++j)
    bPtr[j] = Bm + (n0 + (long long)(j * 16 + col)) * (long long)K + hf * 16;

  v8f acc[4][4];
#pragma unroll
  for (int i = 0; i < 4; ++i)
#pragma unroll
    for (int j = 0; j < 4; ++j) acc[i][j] = (v8f)(0.0f);

  Frags f0, f1;
  load_frags(f0, aPtr, bPtr, 0);

  int k0 = 0;
  for (; k0 < K - 64; k0 += 64) {
    load_frags(f1, aPtr, bPtr, k0 + 32);
    mma_tile(acc, f0);
    load_frags(f0, aPtr, bPtr, k0 + 64);
    mma_tile(acc, f1);
  }
  // Tail: K % 64 == 0, so exactly one double-step remains.
  load_frags(f1, aPtr, bPtr, K - 32);
  mma_tile(acc, f0);
  mma_tile(acc, f1);

  // Epilogue. C fragment layout: VGPR r -> M = 8*hf + r, lane col -> N.
#pragma unroll
  for (int i = 0; i < 4; ++i)
#pragma unroll
    for (int j = 0; j < 4; ++j)
#pragma unroll
      for (int r = 0; r < 8; ++r) {
        long long row = m0 + i * 16 + hf * 8 + r;
        long long cc  = n0 + j * 16 + col;
        float v = acc[i][j][r];
        if (RELU) v = fmaxf(v, 0.0f);
        C[row * (long long)N + cc] = (OUT_T)v;
      }
}

// ---------------------------------------------------------------------------
// Host launcher.
// Inputs: x [B,S,D] f32, theta [D] f32, W1 [F,D] f32, W2 [D,F] f32.
// Output: [B,S,D] f32.  Workspace: Qbf (M*D) | W1bf (F*D) | W2bf (D*F) |
// Hbf (M*F), all bf16.  M=16384, D=1024, F=4096 -> ~176 MB.
// ---------------------------------------------------------------------------
extern "C" void kernel_launch(void* const* d_in, const int* in_sizes, int n_in,
                              void* d_out, int out_size, void* d_ws,
                              size_t ws_size, hipStream_t stream) {
  const float* x     = (const float*)d_in[0];
  const float* theta = (const float*)d_in[1];
  const float* W1    = (const float*)d_in[2];
  const float* W2    = (const float*)d_in[3];
  float* out = (float*)d_out;

  const int D = in_sizes[1];
  const int M = in_sizes[0] / D;           // B*S = 16384
  const int F = in_sizes[2] / D;           // 4096

  char* ws = (char*)d_ws;
  __bf16* qbf  = (__bf16*)ws;
  __bf16* w1bf = qbf + (long long)M * D;
  __bf16* w2bf = w1bf + (long long)F * D;
  __bf16* hbf  = w2bf + (long long)D * F;

  // Pass 1: quantum feature map -> bf16
  long long nq8 = (long long)M * D / 8;
  quantum_pack_bf16<<<(int)((nq8 + 255) / 256), 256, 0, stream>>>(
      x, theta, qbf, nq8, D);

  // Weight conversion (F*D == D*F elements each)
  long long nw8 = (long long)F * D / 8;
  cvt_f32_to_bf16<<<(int)((nw8 + 255) / 256), 256, 0, stream>>>(W1, w1bf, nw8);
  cvt_f32_to_bf16<<<(int)((nw8 + 255) / 256), 256, 0, stream>>>(W2, w2bf, nw8);

  // GEMM1: H = relu(Q @ W1^T)   [M,F], bf16 out
  int tiles1 = (M / 64) * (F / 64);
  gemm_bf16_wmma<true, __bf16><<<tiles1 / 8, 256, 0, stream>>>(
      qbf, w1bf, hbf, M, F, D);

  // GEMM2: out = H @ W2^T       [M,D], f32 out
  int tiles2 = (M / 64) * (D / 64);
  gemm_bf16_wmma<false, float><<<tiles2 / 8, 256, 0, stream>>>(
      hbf, w2bf, out, M, D, F);
}
